// SwinBlock3D_49254684951131
// MI455X (gfx1250) — matrix operand order repaired and checked by reading, not compiled, not run on hardware
//
#include <hip/hip_runtime.h>
#include <hip/hip_bf16.h>
#include <cstdint>

typedef __attribute__((ext_vector_type(16))) _Float16 v16h;
typedef __attribute__((ext_vector_type(8)))  _Float16 v8h;
typedef __attribute__((ext_vector_type(4)))  _Float16 v4h;
typedef __attribute__((ext_vector_type(8)))  float    v8f;

#define WMMA_F16(a,b,c) __builtin_amdgcn_wmma_f32_16x16x32_f16(false,(a),false,(b),(short)0,(c),false,false)

__device__ __forceinline__ v8h ld8h(const _Float16* p){ return *(const v8h*)p; }

__device__ __forceinline__ v16h cat16(v8h a, v8h b){
  v16h r;
#pragma unroll
  for (int i = 0; i < 8; ++i){ r[i] = a[i]; r[8+i] = b[i]; }
  return r;
}

// A operand (16x32 f16): row-major [M x ld] matrix, tile origin (row0, k0).
// Lane l: m = l&15; hi = l>>4. elems 0-7 = K (hi*8 .. +7), elems 8-15 = K (16+hi*8 ..).
__device__ __forceinline__ v16h loadA(const _Float16* base, int ld, int row0, int k0, int lane){
  int hi = lane >> 4, m = lane & 15;
  const _Float16* p = base + (size_t)(row0 + m) * ld + k0;
  return cat16(ld8h(p + hi*8), ld8h(p + 16 + hi*8));
}

// B operand (32x16 f16): transposed weight wT row-major [N x ld] (row = out-ch).
// Lane l: n = l&15; hi = l>>4. elems 0-7 = K (hi*8 .. +7), elems 8-15 = K (16+hi*8 ..).
__device__ __forceinline__ v16h loadB(const _Float16* base, int ld, int n0, int k0, int lane){
  int hi = lane >> 4, n = lane & 15;
  const _Float16* p = base + (size_t)(n0 + n) * ld + k0;
  return cat16(ld8h(p + hi*8), ld8h(p + 16 + hi*8));
}

// Async global->LDS 16B copy per lane (ASYNCcnt path, no VGPR round trip).
__device__ __forceinline__ void async_copy_b128(const _Float16* gsrc, _Float16* ldst){
  unsigned lds_off = (unsigned)(uintptr_t)ldst;   // LDS aperture: low 32 bits = byte offset
  asm volatile("global_load_async_to_lds_b128 %0, %1, off"
               :: "v"(lds_off), "v"(gsrc) : "memory");
}
__device__ __forceinline__ void async_wait(){
  asm volatile("s_wait_asynccnt 0x0" ::: "memory");
}

// -------------------- weight prep: f32 -> f16 transposed --------------------
__global__ void prep_weights(const float* wqkv, const float* wproj,
                             const float* wfc1, const float* wfc2,
                             _Float16* wqkvT, _Float16* wprojT,
                             _Float16* wfc1T, _Float16* wfc2T){
  int idx = blockIdx.x * blockDim.x + threadIdx.x;
  const int S0 = 384*128, S1 = 128*128, S2 = 512*128, S3 = 128*512;
  if (idx < S0){
    int n = idx >> 7, k = idx & 127;
    wqkvT[idx] = (_Float16)wqkv[k*384 + n];
  } else if (idx < S0 + S1){
    int t = idx - S0; int n = t >> 7, k = t & 127;
    wprojT[t] = (_Float16)wproj[k*128 + n];
  } else if (idx < S0 + S1 + S2){
    int t = idx - S0 - S1; int n = t >> 7, k = t & 127;
    wfc1T[t] = (_Float16)wfc1[k*512 + n];
  } else if (idx < S0 + S1 + S2 + S3){
    int t = idx - S0 - S1 - S2; int n = t >> 9, k = t & 511;
    wfc2T[t] = (_Float16)wfc2[k*128 + n];
  }
}

// -------------------- relative position bias MLP --------------------
__global__ void bias_kernel(const float* w1, const float* b1,
                            const float* w2, const float* b2, float* biasT){
  int idx = blockIdx.x * blockDim.x + threadIdx.x;
  if (idx >= 4096) return;
  int n = idx >> 6, m = idx & 63;
  float r0 = (float)((n >> 4)       - (m >> 4));
  float r1 = (float)(((n >> 2) & 3) - ((m >> 2) & 3));
  float r2 = (float)((n & 3)        - (m & 3));
  float o[8];
#pragma unroll
  for (int h = 0; h < 8; ++h) o[h] = b2[h];
  for (int j = 0; j < 64; ++j){
    float hv = fmaxf(r0*w1[j] + r1*w1[64 + j] + r2*w1[128 + j] + b1[j], 0.f);
#pragma unroll
    for (int h = 0; h < 8; ++h) o[h] += hv * w2[j*8 + h];
  }
#pragma unroll
  for (int h = 0; h < 8; ++h) biasT[h*4096 + idx] = o[h];
}

// -------------------- LayerNorm: one wave per token (wave32) --------------
__global__ void ln_kernel(const float* __restrict__ in, const float* __restrict__ gamma,
                          const float* __restrict__ beta, _Float16* __restrict__ out,
                          int remap){
  int wave = threadIdx.x >> 5, lane = threadIdx.x & 31;
  int tok  = blockIdx.x * 8 + wave;
  int c    = lane * 4;
  float4 v = *(const float4*)(in + (size_t)tok*128 + c);
  float s = v.x + v.y + v.z + v.w;
#pragma unroll
  for (int m = 1; m <= 16; m <<= 1) s += __shfl_xor(s, m, 32);
  float mu = s * (1.f/128.f);
  float d0 = v.x - mu, d1 = v.y - mu, d2 = v.z - mu, d3 = v.w - mu;
  float q = d0*d0 + d1*d1 + d2*d2 + d3*d3;
#pragma unroll
  for (int m = 1; m <= 16; m <<= 1) q += __shfl_xor(q, m, 32);
  float inv = rsqrtf(q * (1.f/128.f) + 1e-5f);
  float4 g4 = *(const float4*)(gamma + c);
  float4 b4 = *(const float4*)(beta + c);
  v4h y;
  y[0] = (_Float16)(d0*inv*g4.x + b4.x);
  y[1] = (_Float16)(d1*inv*g4.y + b4.y);
  y[2] = (_Float16)(d2*inv*g4.z + b4.z);
  y[3] = (_Float16)(d3*inv*g4.w + b4.w);

  size_t dst;
  if (remap){
    int b_ = tok >> 15, s_ = tok & 32767;
    int ip = s_ >> 10, jp = (s_ >> 5) & 31, kp = s_ & 31;
    int i = (ip + 30) & 31, j = (jp + 30) & 31, k = (kp + 30) & 31;  // roll -2
    int w = ((b_*8 + (i >> 2))*8 + (j >> 2))*8 + (k >> 2);
    int t = ((i & 3) << 4) | ((j & 3) << 2) | (k & 3);
    dst = ((size_t)w*64 + t)*128;
  } else {
    dst = (size_t)tok*128;
  }
  *(v4h*)(out + dst + c) = y;
}

// -------------------- fused windowed attention --------------------
// 1 block per window (2048), 256 threads = 8 waves.
// dyn LDS (halves): As/ao[64][128] | qs[64][128] | ks[64][128] | vT[128][64] | Ps[8][64][64]
__global__ void attn_kernel(const _Float16* __restrict__ hwin,
                            const _Float16* __restrict__ wqkvT, const float* __restrict__ bqkv,
                            const _Float16* __restrict__ wprojT, const float* __restrict__ bproj,
                            const float* __restrict__ biasT,
                            const float* __restrict__ x, float* __restrict__ x1){
  extern __shared__ _Float16 lds[];
  _Float16* As = lds;                 // staging for h tile, reused as attention-out (ao)
  _Float16* qs = lds + 8192;
  _Float16* ks = lds + 16384;
  _Float16* vT = lds + 24576;         // [128][64]
  _Float16* Ps = lds + 32768;         // [8][64][64]
  _Float16* ao = As;                  // reuse after QKV stage

  int w    = blockIdx.x;
  int tid  = threadIdx.x;
  int wave = tid >> 5, lane = tid & 31;
  int hi   = lane >> 4, ln = lane & 15;

  // prefetch L2-resident weights (global_prefetch_b8)
  __builtin_prefetch(wqkvT + tid*192, 0, 1);
  __builtin_prefetch(wprojT + tid*64, 0, 1);

  // stage h tile (64x128 f16 = 16KB) into LDS via async global->LDS copies
  {
    const _Float16* src = hwin + (size_t)w * 64 * 128 + tid*8;
#pragma unroll
    for (int i = 0; i < 4; ++i)
      async_copy_b128(src + i*2048, As + tid*8 + i*2048);
    async_wait();
  }
  __syncthreads();

  // ---- QKV GEMM: (64x128) @ (128x384), tiles 4 x 24 ----
  for (int tile = wave; tile < 96; tile += 8){
    int mt = tile / 24, nt = tile % 24;
    v8f acc = {};
#pragma unroll
    for (int kk = 0; kk < 4; ++kk){
      v16h a = loadA(As,    128, mt*16, kk*32, lane);
      v16h b = loadB(wqkvT, 128, nt*16, kk*32, lane);
      acc = WMMA_F16(a, b, acc);
    }
    int c = nt*16 + ln;
    float bc = bqkv[c];
    // segment is uniform per tile (nt fixed for the wave this iteration)
    if (nt < 8){
#pragma unroll
      for (int r = 0; r < 8; ++r){
        int m = mt*16 + r + hi*8;
        qs[m*128 + c] = (_Float16)(acc[r] + bc);
      }
    } else if (nt < 16){
      int ck = c - 128;
#pragma unroll
      for (int r = 0; r < 8; ++r){
        int m = mt*16 + r + hi*8;
        ks[m*128 + ck] = (_Float16)(acc[r] + bc);
      }
    } else {
      int cv = c - 256;
#pragma unroll
      for (int r = 0; r < 8; ++r){
        int m = mt*16 + r + hi*8;
        vT[cv*64 + m] = (_Float16)(acc[r] + bc);   // vT[head*16+hd][tok]
      }
    }
  }
  __syncthreads();

  // ---- per-head attention: wave == head ----
  int h = wave;
  const _Float16* qh = qs + h*16;        // row stride 128, HD=16
  const _Float16* kh = ks + h*16;
  const _Float16* vh = vT + h*16*64;     // [16][64]
  _Float16* Ph = Ps + h*64*64;           // [64][64]
  const float* bh = biasT + h*64*64;

  for (int mi = 0; mi < 4; ++mi){
    v8h a0 = ld8h(qh + (size_t)(mi*16 + ln)*128 + hi*8);
    v8h zz = {};
    v16h a = cat16(a0, zz);                // K=16 zero-padded to 32
    v8f accs[4];
#pragma unroll
    for (int j = 0; j < 4; ++j){
      const _Float16* kp = kh + (size_t)(j*16 + ln)*128;
      v16h b = cat16(ld8h(kp + hi*8), zz);             // K hi*8..+7 real, rest zero
      v8f c0 = {};
      accs[j] = WMMA_F16(a, b, c0);
    }
#pragma unroll
    for (int r = 0; r < 8; ++r){
      int nq = mi*16 + r + hi*8;
      float v0 = accs[0][r]*0.25f + bh[nq*64 +  0 + ln];
      float v1 = accs[1][r]*0.25f + bh[nq*64 + 16 + ln];
      float v2 = accs[2][r]*0.25f + bh[nq*64 + 32 + ln];
      float v3 = accs[3][r]*0.25f + bh[nq*64 + 48 + ln];
      float mx = fmaxf(fmaxf(v0, v1), fmaxf(v2, v3));
#pragma unroll
      for (int msk = 1; msk <= 8; msk <<= 1) mx = fmaxf(mx, __shfl_xor(mx, msk, 32));
      float e0 = __expf(v0 - mx), e1 = __expf(v1 - mx);
      float e2 = __expf(v2 - mx), e3 = __expf(v3 - mx);
      float s = e0 + e1 + e2 + e3;
#pragma unroll
      for (int msk = 1; msk <= 8; msk <<= 1) s += __shfl_xor(s, msk, 32);
      float inv = 1.0f / s;
      Ph[nq*64 +  0 + ln] = (_Float16)(e0*inv);
      Ph[nq*64 + 16 + ln] = (_Float16)(e1*inv);
      Ph[nq*64 + 32 + ln] = (_Float16)(e2*inv);
      Ph[nq*64 + 48 + ln] = (_Float16)(e3*inv);
    }
  }

  // ---- out_h = P(64x64) @ V(64x16); write into ao (reused As region) ----
  for (int mi = 0; mi < 4; ++mi){
    v8f acc = {};
#pragma unroll
    for (int kk = 0; kk < 2; ++kk){
      v16h a = loadA(Ph, 64, mi*16, kk*32, lane);
      v16h b = loadB(vh, 64, 0,     kk*32, lane);
      acc = WMMA_F16(a, b, acc);
    }
#pragma unroll
    for (int r = 0; r < 8; ++r){
      int m = mi*16 + r + hi*8;
      ao[m*128 + h*16 + ln] = (_Float16)acc[r];
    }
  }
  __syncthreads();

  // ---- proj (64x128)@(128x128) + inverse roll scatter + residual ----
  int wd = w & 7, wwi = (w >> 3) & 7, wh = (w >> 6) & 7, b_ = w >> 9;
  for (int tile = wave; tile < 32; tile += 8){
    int mt = tile >> 3, nt = tile & 7;
    v8f acc = {};
#pragma unroll
    for (int kk = 0; kk < 4; ++kk){
      v16h a = loadA(ao,     128, mt*16, kk*32, lane);
      v16h b = loadB(wprojT, 128, nt*16, kk*32, lane);
      acc = WMMA_F16(a, b, acc);
    }
    int c = nt*16 + ln;
    float bc = bproj[c];
#pragma unroll
    for (int r = 0; r < 8; ++r){
      int t  = mt*16 + r + hi*8;
      int ti = t >> 4, tj = (t >> 2) & 3, tk = t & 3;
      int i = ((wh*4 + ti) + 2) & 31;      // roll +2
      int j = ((wwi*4 + tj) + 2) & 31;
      int k = ((wd*4 + tk) + 2) & 31;
      size_t idx = ((((size_t)b_*32 + i)*32 + j)*32 + k)*128 + c;
      x1[idx] = x[idx] + acc[r] + bc;
    }
  }
}

// -------------------- fused MLP --------------------
// 1 block per 64-token tile (2048), 256 threads.
// dyn LDS (halves): As[64][128] | hid[64][512] = 40960 halves = 80KB
__global__ void mlp_kernel(const _Float16* __restrict__ ln2,
                           const _Float16* __restrict__ wfc1T, const float* __restrict__ bfc1,
                           const _Float16* __restrict__ wfc2T, const float* __restrict__ bfc2,
                           const float* __restrict__ x1, float* __restrict__ out){
  extern __shared__ _Float16 smem[];
  _Float16* As  = smem;          // 64 x 128
  _Float16* hid = smem + 8192;   // 64 x 512
  int blk  = blockIdx.x;
  int tid  = threadIdx.x;
  int wave = tid >> 5, lane = tid & 31;
  int hi   = lane >> 4, ln = lane & 15;

  __builtin_prefetch(wfc1T + tid*256, 0, 1);
  __builtin_prefetch(wfc2T + tid*256, 0, 1);

  {
    const _Float16* src = ln2 + (size_t)blk * 64 * 128 + tid*8;
#pragma unroll
    for (int i = 0; i < 4; ++i)
      async_copy_b128(src + i*2048, As + tid*8 + i*2048);
    async_wait();
  }
  __syncthreads();

  // fc1 + exact GELU: (64x128)@(128x512), tiles 4 x 32
  for (int tile = wave; tile < 128; tile += 8){
    int mt = tile >> 5, nt = tile & 31;
    v8f acc = {};
#pragma unroll
    for (int kk = 0; kk < 4; ++kk){
      v16h a = loadA(As,    128, mt*16, kk*32, lane);
      v16h b = loadB(wfc1T, 128, nt*16, kk*32, lane);
      acc = WMMA_F16(a, b, acc);
    }
    int c = nt*16 + ln;
    float bc = bfc1[c];
#pragma unroll
    for (int r = 0; r < 8; ++r){
      float vv = acc[r] + bc;
      float g  = 0.5f * vv * (1.0f + erff(vv * 0.70710678118654752f));
      hid[(mt*16 + r + hi*8)*512 + c] = (_Float16)g;
    }
  }
  __syncthreads();

  // fc2 + residual: (64x512)@(512x128), tiles 4 x 8, 16 K steps
  for (int tile = wave; tile < 32; tile += 8){
    int mt = tile >> 3, nt = tile & 7;
    v8f acc = {};
#pragma unroll
    for (int kk = 0; kk < 16; ++kk){
      v16h a = loadA(hid,   512, mt*16, kk*32, lane);
      v16h b = loadB(wfc2T, 512, nt*16, kk*32, lane);
      acc = WMMA_F16(a, b, acc);
    }
    int c = nt*16 + ln;
    float bc = bfc2[c];
#pragma unroll
    for (int r = 0; r < 8; ++r){
      size_t idx = ((size_t)blk*64 + mt*16 + r + hi*8)*128 + c;
      out[idx] = x1[idx] + acc[r] + bc;
    }
  }
}

// -------------------- host launcher --------------------
extern "C" void kernel_launch(void* const* d_in, const int* in_sizes, int n_in,
                              void* d_out, int out_size, void* d_ws, size_t ws_size,
                              hipStream_t stream) {
  const float* x      = (const float*)d_in[0];
  const float* gamma1 = (const float*)d_in[1];
  const float* beta1  = (const float*)d_in[2];
  const float* w_qkv  = (const float*)d_in[3];
  const float* b_qkv  = (const float*)d_in[4];
  const float* w_proj = (const float*)d_in[5];
  const float* b_proj = (const float*)d_in[6];
  const float* w_b1   = (const float*)d_in[7];
  const float* b_b1   = (const float*)d_in[8];
  const float* w_b2   = (const float*)d_in[9];
  const float* b_b2   = (const float*)d_in[10];
  const float* gamma2 = (const float*)d_in[11];
  const float* beta2  = (const float*)d_in[12];
  const float* w_fc1  = (const float*)d_in[13];
  const float* b_fc1  = (const float*)d_in[14];
  const float* w_fc2  = (const float*)d_in[15];
  const float* b_fc2  = (const float*)d_in[16];

  const size_t T = 131072;   // B * 32^3 tokens
  char* ws = (char*)d_ws;
  size_t o = 0;
  float*     x1     = (float*)(ws + o);     o += T*128*sizeof(float);
  _Float16*  hwin   = (_Float16*)(ws + o);  o += T*128*sizeof(_Float16);  // reused as ln2
  float*     biasT  = (float*)(ws + o);     o += 8*64*64*sizeof(float);
  _Float16*  wqkvT  = (_Float16*)(ws + o);  o += 384*128*sizeof(_Float16);
  _Float16*  wprojT = (_Float16*)(ws + o);  o += 128*128*sizeof(_Float16);
  _Float16*  wfc1T  = (_Float16*)(ws + o);  o += 512*128*sizeof(_Float16);
  _Float16*  wfc2T  = (_Float16*)(ws + o);  o += 128*512*sizeof(_Float16);

  prep_weights<<<768, 256, 0, stream>>>(w_qkv, w_proj, w_fc1, w_fc2,
                                        wqkvT, wprojT, wfc1T, wfc2T);
  bias_kernel<<<16, 256, 0, stream>>>(w_b1, b_b1, w_b2, b_b2, biasT);
  ln_kernel<<<16384, 256, 0, stream>>>(x, gamma1, beta1, hwin, 1);
  attn_kernel<<<2048, 256, 131072, stream>>>(hwin, wqkvT, b_qkv, wprojT, b_proj,
                                             biasT, x, x1);
  ln_kernel<<<16384, 256, 0, stream>>>(x1, gamma2, beta2, hwin, 0);
  mlp_kernel<<<2048, 256, 81920, stream>>>(hwin, wfc1T, b_fc1, wfc2T, b_fc2,
                                           x1, (float*)d_out);
}
